// DenseFeatureNumericEmbedding_67104569033282
// MI455X (gfx1250) — compile-verified, matrix operand order checked by reading, not run on hardware
//
#include <hip/hip_runtime.h>

typedef __attribute__((ext_vector_type(16))) _Float16 v16h;
typedef __attribute__((ext_vector_type(2)))  _Float16 h2;
typedef __attribute__((ext_vector_type(8)))  float    v8f;

#define B_   16384
#define F_   128
#define H_   64
#define E_   16
#define WAVES 8
#define TPW   8   // 16-row batch tiles per wave

__global__ __launch_bounds__(256)
void dfne_wmma_kernel(const float* __restrict__ x,
                      const float* __restrict__ W1,
                      const float* __restrict__ b1,
                      const float* __restrict__ W2,
                      const float* __restrict__ b2,
                      float* __restrict__ out)
{
    __shared__ float sW1[H_];
    __shared__ float sB1[H_];
    __shared__ float sW2[E_ * H_];
    __shared__ float sB2[E_];

    const int f   = blockIdx.y;
    const int tid = threadIdx.x;

    // Stage per-feature params in LDS (once per block).
    for (int i = tid; i < H_; i += 256) {
        sW1[i] = W1[f * H_ + i];
        sB1[i] = b1[f * H_ + i];
    }
    for (int i = tid; i < E_ * H_; i += 256)
        sW2[i] = W2[f * E_ * H_ + i];
    if (tid < E_)
        sB2[tid] = b2[f * E_ + tid];
    __syncthreads();

    const int lane = tid & 31;
    const int wave = tid >> 5;
    const int n    = lane & 15;   // A: row m; B: col n; C/D: col n
    const int g    = lane >> 4;   // half-wave group selects K range

    // Bake B matrix (W2[f]^T, WMMA 16-bit 32x16 layout) and packed-f16
    // per-slot-pair W1/b1 into registers once per wave.
    // Slot j -> K mapping: k(j) = (j<8 ? j : j+8) + 8*g ; pairs (2v,2v+1)
    // hit consecutive K, so h2 packing is free.
    h2   w1p[16], b1p[16];
    v16h bm0, bm1;
#pragma unroll
    for (int v = 0; v < 8; ++v) {
        const int j = 2 * v;
        const int k = ((j < 8) ? j : (j + 8)) + 8 * g;
        w1p[v]     = (h2){(_Float16)sW1[k],      (_Float16)sW1[k + 1]};
        w1p[v + 8] = (h2){(_Float16)sW1[k + 32], (_Float16)sW1[k + 33]};
        b1p[v]     = (h2){(_Float16)sB1[k],      (_Float16)sB1[k + 1]};
        b1p[v + 8] = (h2){(_Float16)sB1[k + 32], (_Float16)sB1[k + 33]};
        bm0[j]     = (_Float16)sW2[n * H_ + k];
        bm0[j + 1] = (_Float16)sW2[n * H_ + k + 1];
        bm1[j]     = (_Float16)sW2[n * H_ + k + 32];
        bm1[j + 1] = (_Float16)sW2[n * H_ + k + 33];
    }
    const float bias = sB2[n];

    const long rowBase = (long)blockIdx.x * (WAVES * TPW * 16) + (long)wave * (TPW * 16);
    const h2 zero = (h2){(_Float16)0.0f, (_Float16)0.0f};

#pragma unroll 2
    for (int t = 0; t < TPW; ++t) {
        const long tileRow = rowBase + (long)t * 16;

        // x value for this lane's A row (m = lane & 15), broadcast to h2
        const float    xv = x[(tileRow + n) * F_ + f];
        const _Float16 xh = (_Float16)xv;
        const h2       x2 = (h2){xh, xh};

        // A = relu(x*W1 + b1) computed directly in packed f16
        // (v_pk_fma_f16 + v_pk_max_num_f16), 16x32 WMMA layout.
        union { v16h v; h2 p[8]; } A0, A1;
#pragma unroll
        for (int v = 0; v < 8; ++v) {
            h2 t0 = x2 * w1p[v]     + b1p[v];
            h2 t1 = x2 * w1p[v + 8] + b1p[v + 8];
            A0.p[v] = __builtin_elementwise_max(t0, zero);
            A1.p[v] = __builtin_elementwise_max(t1, zero);
        }

        // C starts at the bias (C/D layout: column n fixed per lane)
        v8f c;
#pragma unroll
        for (int r = 0; r < 8; ++r) c[r] = bias;

        c = __builtin_amdgcn_wmma_f32_16x16x32_f16(false, A0.v, false, bm0,
                                                   (short)0, c, false, false);
        c = __builtin_amdgcn_wmma_f32_16x16x32_f16(false, A1.v, false, bm1,
                                                   (short)0, c, false, false);

        // D element (m, n): vgpr r -> m = r + 8*g ; store 16x16 f32 tile.
        float* o = out + (tileRow + 8 * g) * (long)(F_ * E_) + (long)f * E_ + n;
#pragma unroll
        for (int r = 0; r < 8; ++r)
            o[(long)r * (F_ * E_)] = c[r];
    }
}

extern "C" void kernel_launch(void* const* d_in, const int* in_sizes, int n_in,
                              void* d_out, int out_size, void* d_ws, size_t ws_size,
                              hipStream_t stream) {
    const float* x  = (const float*)d_in[0];
    const float* W1 = (const float*)d_in[1];
    const float* b1 = (const float*)d_in[2];
    const float* W2 = (const float*)d_in[3];
    const float* b2 = (const float*)d_in[4];
    float* out = (float*)d_out;

    dim3 grid(B_ / (WAVES * TPW * 16), F_);  // (16, 128)
    dim3 block(256);
    dfne_wmma_kernel<<<grid, block, 0, stream>>>(x, W1, b1, W2, b2, out);
}
